// MixedScore_MultiHeadAttention_45380624450006
// MI455X (gfx1250) — matrix operand order repaired and checked by reading, not compiled, not run on hardware
//
#include <hip/hip_runtime.h>

// MixedScore MultiHeadAttention, fused, MI455X (gfx1250, wave32).
// B=2 H=8 N=512 D=16 E=128.  One workgroup per (b, n) query row;
// 8 waves == 8 heads.  z is streamed exactly once (268 MB => ~11.5us floor
// at 23.3 TB/s); the z @ [Wz1|Wz2] projection runs on v_wmma_f32_16x16x32_bf16.
// Software-pipelined at two levels:
//  - global: z tile t+1 loads are in flight while tile t computes
//  - LDS:    A-fragments ping-pong between two 8-VGPR buffers so ds_load
//            latency overlaps the WMMA chain instead of draining per K-step.

#define B_ 2
#define H_ 8
#define N_ 512
#define D_ 16
#define E_ 128
#define MT 16                 // m-tile
#define TILES (N_ / MT)

typedef __attribute__((ext_vector_type(16))) __bf16 v16bf;
typedef __attribute__((ext_vector_type(8)))  float  v8f;

union AF { v16bf v; uint4 q4[2]; };
union BFr { v16bf v; unsigned u[8]; };

static __device__ __forceinline__ unsigned short f2bf(float f) {
  unsigned u = __builtin_bit_cast(unsigned, f);
  unsigned r = 0x7FFFu + ((u >> 16) & 1u);   // round-to-nearest-even
  return (unsigned short)((u + r) >> 16);
}
static __device__ __forceinline__ unsigned packbf(float a, float b) {
  return (unsigned)f2bf(a) | ((unsigned)f2bf(b) << 16);
}

__global__ __launch_bounds__(256)
void fused_mixedscore_attn(const float* __restrict__ q,
                           const float* __restrict__ k,
                           const float* __restrict__ v,
                           const float* __restrict__ z,
                           const float* __restrict__ Wz1,
                           const float* __restrict__ Wz2,
                           float* __restrict__ out)
{
  __shared__ unsigned short zlds[MT][E_];        // 4 KB  bf16 z tile (16 x 128)
  __shared__ float zproj[MT][2 * H_ * D_];       // 16 KB z1|z2 projected tile
  __shared__ float score_s[H_][N_];              // 16 KB per-head score rows

  const int tid  = threadIdx.x;
  const int w    = tid >> 5;        // wave id == head id
  const int lane = tid & 31;
  const int b    = blockIdx.x / N_;
  const int nrow = blockIdx.x - b * N_;

  const int half = lane >> 4;       // 0: lanes 0-15, 1: lanes 16-31
  const int nloc = lane & 15;       // N-col / m-row within tile

  // ---- B fragments of [Wz1|Wz2] in bf16, held in registers -----------------
  // B 32x16, contiguous K order (same as A, not the ISA table order): lane = col n, half selects K 0-15 / 16-31,
  // dword r holds K pair (2r, 2r+1).
  BFr bfrag[2][4];
  const int ct0 = w * 2;            // this wave's first 16-channel tile (0..14)
  #pragma unroll
  for (int c = 0; c < 2; ++c) {
    const int ch = (ct0 + c) * 16 + nloc;                 // 0..255
    const float* W = (ch < H_ * D_) ? (Wz1 + ch) : (Wz2 + (ch - H_ * D_));
    #pragma unroll
    for (int kk = 0; kk < 4; ++kk) {
      #pragma unroll
      for (int r = 0; r < 8; ++r) {
        const int k0 = kk * 32 + half * 16 + 2 * r;       // K index in 0..127
        bfrag[c][kk].u[r] = packbf(W[(size_t)k0 * (H_ * D_)],
                                   W[(size_t)(k0 + 1) * (H_ * D_)]);
      }
    }
  }

  // ---- q registers for the score phase (head w, row nrow, 8 d values) -----
  const int d0 = half * 8;
  float qv[8];
  {
    const float* qp = q + (((size_t)b * H_ + w) * N_ + nrow) * D_ + d0;
    #pragma unroll
    for (int j = 0; j < 8; ++j) qv[j] = qp[j];
  }

  const float* zrow = z + (((size_t)b * N_ + nrow) * N_) * E_;
  const float* krow = k + (((size_t)b * H_ + w) * N_) * D_;

  const int ml_st = tid >> 4;             // staging row   (tid*8)/128
  const int e_st  = (tid * 8) & 127;      // staging column

  // ---- software pipeline: preload tile 0 into registers --------------------
  float4 zf0 = *(const float4*)(zrow + (size_t)tid * 8);
  float4 zf1 = *(const float4*)(zrow + (size_t)tid * 8 + 4);

  for (int t = 0; t < TILES; ++t) {
    const int m0 = t * MT;

    // ---- convert in-flight tile regs -> bf16 in LDS -----------------------
    {
      unsigned* dst = (unsigned*)&zlds[ml_st][e_st];
      dst[0] = packbf(zf0.x, zf0.y);
      dst[1] = packbf(zf0.z, zf0.w);
      dst[2] = packbf(zf1.x, zf1.y);
      dst[3] = packbf(zf1.z, zf1.w);
    }
    // ---- issue global loads for tile t+1 (land next iteration) ------------
    if (t + 1 < TILES) {
      const float* src = zrow + (size_t)(m0 + MT) * E_ + tid * 8;
      zf0 = *(const float4*)(src);
      zf1 = *(const float4*)(src + 4);
      if (t + 2 < TILES)
        __builtin_prefetch(zrow + (size_t)(m0 + 2 * MT) * E_ + tid * 8, 0, 1);
    }
    __syncthreads();

    // ---- projection: z_tile[16x128]bf16 @ W[128 x 2*16ch] -> zproj --------
    // Explicit ping-pong between two A-fragment buffers: ds_loads for the
    // next K-step issue between the WMMAs of the current one.
    const unsigned short* zr = &zlds[nloc][0];
    AF aA, aB;
    aA.q4[0] = *(const uint4*)(zr + 0 * 32 + half * 16);
    aA.q4[1] = *(const uint4*)(zr + 0 * 32 + 8 + half * 16);
    aB.q4[0] = *(const uint4*)(zr + 1 * 32 + half * 16);
    aB.q4[1] = *(const uint4*)(zr + 1 * 32 + 8 + half * 16);

    v8f acc0 = {}, acc1 = {};
    acc0 = __builtin_amdgcn_wmma_f32_16x16x32_bf16(
               false, aA.v, false, bfrag[0][0].v, (short)0, acc0, false, false);
    acc1 = __builtin_amdgcn_wmma_f32_16x16x32_bf16(
               false, aA.v, false, bfrag[1][0].v, (short)0, acc1, false, false);

    aA.q4[0] = *(const uint4*)(zr + 2 * 32 + half * 16);
    aA.q4[1] = *(const uint4*)(zr + 2 * 32 + 8 + half * 16);

    acc0 = __builtin_amdgcn_wmma_f32_16x16x32_bf16(
               false, aB.v, false, bfrag[0][1].v, (short)0, acc0, false, false);
    acc1 = __builtin_amdgcn_wmma_f32_16x16x32_bf16(
               false, aB.v, false, bfrag[1][1].v, (short)0, acc1, false, false);

    aB.q4[0] = *(const uint4*)(zr + 3 * 32 + half * 16);
    aB.q4[1] = *(const uint4*)(zr + 3 * 32 + 8 + half * 16);

    acc0 = __builtin_amdgcn_wmma_f32_16x16x32_bf16(
               false, aA.v, false, bfrag[0][2].v, (short)0, acc0, false, false);
    acc1 = __builtin_amdgcn_wmma_f32_16x16x32_bf16(
               false, aA.v, false, bfrag[1][2].v, (short)0, acc1, false, false);
    acc0 = __builtin_amdgcn_wmma_f32_16x16x32_bf16(
               false, aB.v, false, bfrag[0][3].v, (short)0, acc0, false, false);
    acc1 = __builtin_amdgcn_wmma_f32_16x16x32_bf16(
               false, aB.v, false, bfrag[1][3].v, (short)0, acc1, false, false);

    // C/D layout: VGPR i -> row (i + 8*half), col = nloc within channel tile.
    #pragma unroll
    for (int i = 0; i < 8; ++i) {
      zproj[i + 8 * half][ct0 * 16 + nloc]       = acc0[i];
      zproj[i + 8 * half][(ct0 + 1) * 16 + nloc] = acc1[i];
    }
    __syncthreads();

    // ---- score: head w, 16 m rows; half-wave splits d into 8+8 ------------
    {
      const int ml = nloc;
      const float* kp = krow + (size_t)(m0 + ml) * D_ + d0;
      float s = 0.f;
      #pragma unroll
      for (int j = 0; j < 8; ++j) {
        const float z1 = zproj[ml][w * D_ + d0 + j];
        const float z2 = zproj[ml][H_ * D_ + w * D_ + d0 + j];
        s += (qv[j] + z1) * (kp[j] + z2);
      }
      s += __shfl_xor(s, 16, 32);
      if (half == 0) score_s[w][m0 + ml] = s * 0.25f;   // 1/sqrt(D)
    }
    __syncthreads();
  }

  // ---- per-head softmax over m, then out = weights @ v ---------------------
  {
    float mx = -3.4e38f;
    #pragma unroll
    for (int i = 0; i < 16; ++i) mx = fmaxf(mx, score_s[w][lane + i * 32]);
    #pragma unroll
    for (int off = 16; off >= 1; off >>= 1) mx = fmaxf(mx, __shfl_xor(mx, off, 32));

    float se = 0.f;
    #pragma unroll
    for (int i = 0; i < 16; ++i) {
      const float e = __expf(score_s[w][lane + i * 32] - mx);
      score_s[w][lane + i * 32] = e;      // store unnormalized weights in place
      se += e;
    }
    #pragma unroll
    for (int off = 16; off >= 1; off >>= 1) se += __shfl_xor(se, off, 32);
    __syncthreads();

    const int d = nloc;                   // lane -> output dim, halves split m
    const float* vrow = v + (((size_t)b * H_ + w) * N_) * D_ + d;
    float acc = 0.f;
    for (int m = half * 256; m < half * 256 + 256; ++m)
      acc += score_s[w][m] * vrow[(size_t)m * D_];
    acc += __shfl_xor(acc, 16, 32);

    if (half == 0)
      out[((size_t)b * N_ + nrow) * (H_ * D_) + w * D_ + d] = acc / se;
  }
}

extern "C" void kernel_launch(void* const* d_in, const int* in_sizes, int n_in,
                              void* d_out, int out_size, void* d_ws, size_t ws_size,
                              hipStream_t stream) {
  (void)in_sizes; (void)n_in; (void)d_ws; (void)ws_size; (void)out_size;
  const float* q   = (const float*)d_in[0];
  const float* k   = (const float*)d_in[1];
  const float* v   = (const float*)d_in[2];
  const float* z   = (const float*)d_in[3];
  const float* Wz1 = (const float*)d_in[4];
  const float* Wz2 = (const float*)d_in[5];
  float* out = (float*)d_out;

  dim3 grid(B_ * N_);
  dim3 block(256);
  fused_mixedscore_attn<<<grid, block, 0, stream>>>(q, k, v, z, Wz1, Wz2, out);
}
